// LinearAttention3D_51866025066537
// MI455X (gfx1250) — matrix operand order repaired and checked by reading, not compiled, not run on hardware
//
#include <hip/hip_runtime.h>

#define NSPAT 32768   // 32*32*32 spatial positions
#define CDIM  256
#define QKVO  384
#define HID   128
#define SCALE 0.17677669529663689f
#define LNEPS 1e-5f
#define PAD   40      // LDS row pad (elements): 80B rows, matches TDM pad (16dw interval + 4dw pad)

typedef __attribute__((ext_vector_type(16))) __bf16 bfx16;
typedef __attribute__((ext_vector_type(8)))  float  fx8;
typedef unsigned int u32x4 __attribute__((ext_vector_type(4)));
typedef int          i32x8 __attribute__((ext_vector_type(8)));
typedef int          i32x4 __attribute__((ext_vector_type(4)));

union BF16x16 { unsigned int u[8]; bfx16 v; };

__device__ __forceinline__ unsigned short f2bf(float f) {
    unsigned int u = __float_as_uint(f);
    unsigned int r = u + 0x7FFFu + ((u >> 16) & 1u);   // round-to-nearest-even
    return (unsigned short)(r >> 16);
}
__device__ __forceinline__ unsigned int pack2bf(float lo, float hi) {
    return (unsigned int)f2bf(lo) | ((unsigned int)f2bf(hi) << 16);
}

// ---- Tensor Data Mover: DMA a (tile_w x tile_h) bf16 tile (row stride in elems) into LDS,
//      inserting 8 pad elements after every 32-element row (pad_interval=16 dwords, pad_amount=4 dwords)
//      so the LDS image lands with pitch PAD=40. 1-row tiles are issued as 1-D transfers.
//      6-arg builtin form (clang-23 / therock-10.0 signature).
__device__ __forceinline__ void tdm_load_bf16(unsigned int lds_off, const void* gptr,
                                              unsigned int tile_w, unsigned int tile_h,
                                              unsigned int row_stride) {
    unsigned long long ga = (unsigned long long)(size_t)gptr;
    u32x4 g0;
    g0.x = 1u;                                               // count=1, user descriptor
    g0.y = lds_off;                                          // LDS byte address
    g0.z = (unsigned int)ga;                                 // global addr [31:0]
    g0.w = (unsigned int)((ga >> 32) & 0x01FFFFFFu) | (2u << 30);  // addr [56:32] | type=2
    i32x8 g1;
    g1[0] = (1 << 16) | (1 << 20) | (3 << 22) | (3 << 25);   // data 2B | pad_en | 16dw interval | 4dw pad
    g1[1] = (int)((tile_w & 0xFFFFu) << 16);                 // tensor_dim0[15:0] (== tile width)
    g1[2] = (int)((tile_h & 0xFFFFu) << 16);                 // tensor_dim1[15:0]
    g1[3] = (int)((tile_w & 0xFFFFu) << 16);                 // tile_dim0
    g1[4] = (int)(tile_h > 1 ? (tile_h & 0xFFFFu) : 0u);     // tile_dim1 (0 => 1-D)
    g1[5] = (int)row_stride;                                 // tensor_dim0_stride[31:0]
    g1[6] = 0;
    g1[7] = 0;
    i32x4 z4 = {0, 0, 0, 0};
    i32x8 z8 = {0, 0, 0, 0, 0, 0, 0, 0};
    __builtin_amdgcn_tensor_load_to_lds(g0, g1, z4, z4, z8, 0);
}

// ---------------------------------------------------------------- plain convert f32 -> bf16 (weights)
__global__ void __launch_bounds__(256) cvt_bf16_kernel(const float* __restrict__ src,
                                                       unsigned short* __restrict__ dst, int n) {
    int i = blockIdx.x * 256 + threadIdx.x;
    int stride = gridDim.x * 256;
    for (; i < n; i += stride) dst[i] = f2bf(src[i]);
}

// ---------------------------------------------------------------- x (f32 [b][c][n]) -> blocked bf16 x_t[b][c/32][n][32]
__global__ void __launch_bounds__(256) cvt_x_blocked_kernel(const float* __restrict__ x,
                                                            unsigned short* __restrict__ xt) {
    __shared__ unsigned short ldsT[256 * PAD];
    const int t = threadIdx.x;
    const int n0 = blockIdx.x * 256, cb = blockIdx.y, b = blockIdx.z;
#pragma unroll
    for (int c = 0; c < 32; ++c)
        ldsT[t * PAD + c] = f2bf(x[((size_t)b * CDIM + cb * 32 + c) * NSPAT + n0 + t]);
    __syncthreads();
    unsigned short* dst = xt + ((size_t)(b * 8 + cb) * NSPAT + n0 + t) * 32;
#pragma unroll
    for (int i = 0; i < 4; ++i)
        *(uint4*)&dst[i * 8] = *(const uint4*)&ldsT[t * PAD + i * 8];
}

// ---------------------------------------------------------------- GEMM1: qkv = w_qkv (384x256) @ x (256 x n), per batch
// grid (NSPAT/64, 384/128, B), block 256 (8 waves, each 32x32 = 2x2 frags)
// TDM double-buffered staging: wave 0 DMAs next k-tile while all waves run WMMAs.
__global__ void __launch_bounds__(256) gemm_qkv_kernel(const unsigned short* __restrict__ wq,
                                                       const unsigned short* __restrict__ xt,
                                                       float* __restrict__ qkv) {
    __shared__ unsigned short ldsA[2][128 * PAD];   // [m][k]
    __shared__ unsigned short ldsB[2][64 * PAD];    // [n][k]
    const int t = threadIdx.x;
    const int L = t & 31, w = t >> 5;
    const int wm = w & 3, wn = w >> 2;              // 4x2 wave grid -> 128M x 64N block tile
    const int b = blockIdx.z;
    const int mblock = blockIdx.y * 128;
    const int n0 = blockIdx.x * 64;
    const int hi = L >> 4;
    fx8 acc[2][2] = {};

    if (w == 0) {                                   // prologue: DMA k-step 0 into buffer 0
        tdm_load_bf16((unsigned int)(size_t)&ldsA[0][0], &wq[mblock * CDIM], 32, 128, CDIM);
        tdm_load_bf16((unsigned int)(size_t)&ldsB[0][0],
                      &xt[((size_t)(b * 8) * NSPAT + n0) * 32], 2048, 1, 2048);
    }
    const int steps = CDIM / 32;                    // 8
    for (int s = 0; s < steps; ++s) {
        const int cur = s & 1;
        __syncthreads();                            // everyone done reading buffer 1-cur
        if (w == 0) {
            if (s + 1 < steps) {
                tdm_load_bf16((unsigned int)(size_t)&ldsA[1 - cur][0],
                              &wq[mblock * CDIM + (s + 1) * 32], 32, 128, CDIM);
                tdm_load_bf16((unsigned int)(size_t)&ldsB[1 - cur][0],
                              &xt[((size_t)(b * 8 + s + 1) * NSPAT + n0) * 32], 2048, 1, 2048);
                __builtin_amdgcn_s_wait_tensorcnt(2);   // pair for step s complete (in-order)
            } else {
                __builtin_amdgcn_s_wait_tensorcnt(0);
            }
        }
        __syncthreads();                            // buffer cur published
        BF16x16 a[2], bb[2];
#pragma unroll
        for (int mi = 0; mi < 2; ++mi) {
            int arow = (wm * 32 + mi * 16 + (L & 15)) * PAD;
#pragma unroll
            for (int j = 0; j < 8; ++j) {
                int ka = 2 * j + hi * 16;
                a[mi].u[j] = *(const unsigned int*)&ldsA[cur][arow + ka];
            }
        }
#pragma unroll
        for (int ni = 0; ni < 2; ++ni) {
            int brow = (wn * 32 + ni * 16 + (L & 15)) * PAD;
#pragma unroll
            for (int j = 0; j < 8; ++j)
                bb[ni].u[j] = *(const unsigned int*)&ldsB[cur][brow + 2 * j + hi * 16];
        }
#pragma unroll
        for (int mi = 0; mi < 2; ++mi)
#pragma unroll
            for (int ni = 0; ni < 2; ++ni)
                acc[mi][ni] = __builtin_amdgcn_wmma_f32_16x16x32_bf16(
                    false, a[mi].v, false, bb[ni].v, (short)0, acc[mi][ni], false, false);
    }
#pragma unroll
    for (int mi = 0; mi < 2; ++mi) {
        int mg = mblock + wm * 32 + mi * 16 + hi * 8;
#pragma unroll
        for (int ni = 0; ni < 2; ++ni) {
            int ncol = n0 + wn * 32 + ni * 16 + (L & 15);
#pragma unroll
            for (int r = 0; r < 8; ++r)
                qkv[((size_t)b * QKVO + mg + r) * NSPAT + ncol] = acc[mi][ni][r];
        }
    }
}

// ---------------------------------------------------------------- k-softmax row stats (max + inv-sum over n)
__global__ void __launch_bounds__(256) krow_stats_kernel(const float* __restrict__ qkv,
                                                         float* __restrict__ rmax,
                                                         float* __restrict__ rinv) {
    __shared__ float sm[256], ss[256];
    const int rid = blockIdx.x;                      // b*128 + h*32 + d  (512 rows)
    const int b = rid >> 7, o = rid & 127;
    const float* row = qkv + ((size_t)b * QKVO + 128 + o) * NSPAT;
    const int t = threadIdx.x;
    float m = -3.4e38f, s = 0.f;
    for (int i = t; i < NSPAT; i += 256) {
        float x = row[i];
        if (x > m) { s = s * __expf(m - x); m = x; }
        s += __expf(x - m);
    }
    sm[t] = m; ss[t] = s;
    __syncthreads();
    for (int off = 128; off > 0; off >>= 1) {
        if (t < off) {
            float m2 = sm[t + off], s2 = ss[t + off];
            float M = fmaxf(sm[t], m2);
            ss[t] = ss[t] * __expf(sm[t] - M) + s2 * __expf(m2 - M);
            sm[t] = M;
        }
        __syncthreads();
    }
    if (t == 0) { rmax[rid] = sm[0]; rinv[rid] = 1.0f / ss[0]; }
}

// ---------------------------------------------------------------- context[d,e] = sum_n softmax(k)[d,n] * v[e,n]
__global__ void __launch_bounds__(128) context_kernel(const float* __restrict__ qkv,
                                                      const float* __restrict__ rmax,
                                                      const float* __restrict__ rinv,
                                                      float* __restrict__ ctx) {
    __shared__ unsigned short kbuf[32 * PAD];
    __shared__ unsigned short vbuf[32 * PAD];
    const int bh = blockIdx.x;
    const int b = bh >> 2, h = bh & 3;
    const int t = threadIdx.x, L = t & 31, w = t >> 5;
    const int dt = w >> 1, et = w & 1;
    const int hi = L >> 4;
    fx8 acc = {};
    const float* kb = qkv + ((size_t)b * QKVO + 128 + h * 32) * NSPAT;
    const float* vb = qkv + ((size_t)b * QKVO + 256 + h * 32) * NSPAT;
    for (int n0 = 0; n0 < NSPAT; n0 += 32) {
#pragma unroll
        for (int i = 0; i < 4; ++i) {                // 512 bf16-pairs per buffer, 4 pairs/thread
            int p = t + 128 * i; int d = p >> 4, nn2 = (p & 15) * 2;
            int rid = bh * 32 + d;
            float2 kx = *(const float2*)&kb[(size_t)d * NSPAT + n0 + nn2];
            float rm = rmax[rid], ri = rinv[rid];
            *(unsigned int*)&kbuf[d * PAD + nn2] =
                pack2bf(__expf(kx.x - rm) * ri, __expf(kx.y - rm) * ri);
            float2 vx = *(const float2*)&vb[(size_t)d * NSPAT + n0 + nn2];
            *(unsigned int*)&vbuf[d * PAD + nn2] = pack2bf(vx.x, vx.y);
        }
        __syncthreads();
        BF16x16 a, bbf;
        const int arow = (dt * 16 + (L & 15)) * PAD;
        const int brow = (et * 16 + (L & 15)) * PAD;
#pragma unroll
        for (int j = 0; j < 8; ++j) {
            int ka = 2 * j + hi * 16;
            a.u[j]   = *(const unsigned int*)&kbuf[arow + ka];
            bbf.u[j] = *(const unsigned int*)&vbuf[brow + 2 * j + hi * 16];
        }
        acc = __builtin_amdgcn_wmma_f32_16x16x32_bf16(false, a.v, false, bbf.v, (short)0, acc, false, false);
        __syncthreads();
    }
    const int d0 = dt * 16 + hi * 8;
    const int e0 = et * 16 + (L & 15);
#pragma unroll
    for (int r = 0; r < 8; ++r)
        ctx[bh * 1024 + (d0 + r) * 32 + e0] = acc[r];
}

// ---------------------------------------------------------------- fused q-softmax + ctx apply -> out_t[b][h][n][32] (bf16 blocked)
__global__ void __launch_bounds__(256) qctx_kernel(const float* __restrict__ qkv,
                                                   const float* __restrict__ ctx,
                                                   unsigned short* __restrict__ outt) {
    __shared__ float c[1024];
    const int bh = blockIdx.y, b = bh >> 2, h = bh & 3;
    const int t = threadIdx.x;
    const int n = blockIdx.x * 256 + t;
#pragma unroll
    for (int i = 0; i < 4; ++i) c[t + 256 * i] = ctx[bh * 1024 + t + 256 * i];
    __syncthreads();
    const float* qb = qkv + ((size_t)b * QKVO + h * 32) * NSPAT + n;
    float q[32];
    float m = -3.4e38f;
#pragma unroll
    for (int d = 0; d < 32; ++d) { q[d] = qb[(size_t)d * NSPAT]; m = fmaxf(m, q[d]); }
    float s = 0.f;
#pragma unroll
    for (int d = 0; d < 32; ++d) { q[d] = __expf(q[d] - m); s += q[d]; }
    const float inv = SCALE / s;
#pragma unroll
    for (int d = 0; d < 32; ++d) q[d] *= inv;
    unsigned int pk[16];
#pragma unroll
    for (int e2 = 0; e2 < 16; ++e2) {
        float a0 = 0.f, a1 = 0.f;
#pragma unroll
        for (int d = 0; d < 32; ++d) {
            a0 += c[d * 32 + 2 * e2]     * q[d];
            a1 += c[d * 32 + 2 * e2 + 1] * q[d];
        }
        pk[e2] = pack2bf(a0, a1);
    }
    unsigned short* dst = outt + ((size_t)(b * 4 + h) * NSPAT + n) * 32;   // 64B contiguous
#pragma unroll
    for (int i = 0; i < 4; ++i)
        *(uint4*)&dst[i * 8] = *(const uint4*)&pk[i * 4];
}

// ---------------------------------------------------------------- GEMM2 (w_out 256x128 @ out 128xn) + bias + LayerNorm fused
// grid (NSPAT/64, B), block 256 (8 waves: each 32M x 64N; full M=256 -> in-block LN). TDM double-buffered.
__global__ void __launch_bounds__(256) gemm_out_ln_kernel(const unsigned short* __restrict__ wo,
                                                          const unsigned short* __restrict__ outt,
                                                          const float* __restrict__ bias,
                                                          const float* __restrict__ gamma,
                                                          const float* __restrict__ beta,
                                                          float* __restrict__ y) {
    __shared__ unsigned short ldsW[2][256 * PAD];
    __shared__ unsigned short ldsX[2][64 * PAD];
    __shared__ float s_sum[64], s_sq[64];
    const int t = threadIdx.x, L = t & 31, w = t >> 5;
    const int b = blockIdx.y;
    const int n0 = blockIdx.x * 64;
    const int hi = L >> 4;
    fx8 acc[2][4] = {};

    if (w == 0) {
        tdm_load_bf16((unsigned int)(size_t)&ldsW[0][0], &wo[0], 32, 256, HID);
        tdm_load_bf16((unsigned int)(size_t)&ldsX[0][0],
                      &outt[((size_t)(b * 4) * NSPAT + n0) * 32], 2048, 1, 2048);
    }
    const int steps = HID / 32;                      // 4
    for (int s = 0; s < steps; ++s) {
        const int cur = s & 1;
        __syncthreads();
        if (w == 0) {
            if (s + 1 < steps) {
                tdm_load_bf16((unsigned int)(size_t)&ldsW[1 - cur][0],
                              &wo[(s + 1) * 32], 32, 256, HID);
                tdm_load_bf16((unsigned int)(size_t)&ldsX[1 - cur][0],
                              &outt[((size_t)(b * 4 + s + 1) * NSPAT + n0) * 32], 2048, 1, 2048);
                __builtin_amdgcn_s_wait_tensorcnt(2);
            } else {
                __builtin_amdgcn_s_wait_tensorcnt(0);
            }
        }
        __syncthreads();
        BF16x16 a[2], bb[4];
#pragma unroll
        for (int mi = 0; mi < 2; ++mi) {
            int arow = (w * 32 + mi * 16 + (L & 15)) * PAD;
#pragma unroll
            for (int j = 0; j < 8; ++j) {
                int ka = 2 * j + hi * 16;
                a[mi].u[j] = *(const unsigned int*)&ldsW[cur][arow + ka];
            }
        }
#pragma unroll
        for (int nt = 0; nt < 4; ++nt) {
            int brow = (nt * 16 + (L & 15)) * PAD;
#pragma unroll
            for (int j = 0; j < 8; ++j)
                bb[nt].u[j] = *(const unsigned int*)&ldsX[cur][brow + 2 * j + hi * 16];
        }
#pragma unroll
        for (int mi = 0; mi < 2; ++mi)
#pragma unroll
            for (int nt = 0; nt < 4; ++nt)
                acc[mi][nt] = __builtin_amdgcn_wmma_f32_16x16x32_bf16(
                    false, a[mi].v, false, bb[nt].v, (short)0, acc[mi][nt], false, false);
    }
    // fused LayerNorm over M=256 channels for the 64 columns
    const int m0 = w * 32 + hi * 8;
    if (t < 64) { s_sum[t] = 0.f; s_sq[t] = 0.f; }
    __syncthreads();
#pragma unroll
    for (int nt = 0; nt < 4; ++nt) {
        float lsum = 0.f, lsq = 0.f;
#pragma unroll
        for (int mi = 0; mi < 2; ++mi)
#pragma unroll
            for (int r = 0; r < 8; ++r) {
                float v = acc[mi][nt][r] + bias[m0 + mi * 16 + r];
                lsum += v; lsq += v * v;
            }
        atomicAdd(&s_sum[nt * 16 + (L & 15)], lsum);   // ds_add_f32
        atomicAdd(&s_sq[nt * 16 + (L & 15)], lsq);
    }
    __syncthreads();
#pragma unroll
    for (int nt = 0; nt < 4; ++nt) {
        int N = nt * 16 + (L & 15);
        float mean = s_sum[N] * (1.0f / 256.0f);
        float var  = s_sq[N] * (1.0f / 256.0f) - mean * mean;
        float rstd = rsqrtf(var + LNEPS);
#pragma unroll
        for (int mi = 0; mi < 2; ++mi)
#pragma unroll
            for (int r = 0; r < 8; ++r) {
                int M = m0 + mi * 16 + r;
                float v = acc[mi][nt][r] + bias[M];
                y[((size_t)b * CDIM + M) * NSPAT + n0 + N] = (v - mean) * rstd * gamma[M] + beta[M];
            }
    }
}

// ---------------------------------------------------------------- launcher
extern "C" void kernel_launch(void* const* d_in, const int* in_sizes, int n_in,
                              void* d_out, int out_size, void* d_ws, size_t ws_size,
                              hipStream_t stream) {
    const float* x     = (const float*)d_in[0];
    const float* wq    = (const float*)d_in[1];
    const float* wo    = (const float*)d_in[2];
    const float* bias  = (const float*)d_in[3];
    const float* gamma = (const float*)d_in[4];
    const float* beta  = (const float*)d_in[5];
    float* y = (float*)d_out;

    char* ws = (char*)d_ws;
    unsigned short* x_t    = (unsigned short*)(ws + 0);            //  4*256*32768*2
    unsigned short* wq_bf  = (unsigned short*)(ws + 67108864);     //  384*256*2
    unsigned short* wo_bf  = (unsigned short*)(ws + 67305472);     //  256*128*2
    float*          qkv    = (float*)(ws + 67371008);              //  4*384*32768*4
    float*          rmax   = (float*)(ws + 268697600);             //  512*4
    float*          rinv   = (float*)(ws + 268699648);             //  512*4
    float*          ctx    = (float*)(ws + 268701696);             //  16*1024*4
    unsigned short* out_t  = (unsigned short*)(ws + 268767232);    //  4*128*32768*2

    cvt_bf16_kernel<<<96, 256, 0, stream>>>(wq, wq_bf, QKVO * CDIM);
    cvt_bf16_kernel<<<32, 256, 0, stream>>>(wo, wo_bf, CDIM * HID);
    cvt_x_blocked_kernel<<<dim3(NSPAT / 256, 8, 4), 256, 0, stream>>>(x, x_t);

    gemm_qkv_kernel<<<dim3(NSPAT / 64, QKVO / 128, 4), 256, 0, stream>>>(wq_bf, x_t, qkv);
    krow_stats_kernel<<<512, 256, 0, stream>>>(qkv, rmax, rinv);
    context_kernel<<<16, 128, 0, stream>>>(qkv, rmax, rinv, ctx);
    qctx_kernel<<<dim3(NSPAT / 256, 16), 256, 0, stream>>>(qkv, ctx, out_t);
    gemm_out_ln_kernel<<<dim3(NSPAT / 64, 4), 256, 0, stream>>>(wo_bf, out_t, bias, gamma, beta, y);
}
